// SampledSymmetrizationMLP_11982958756741
// MI455X (gfx1250) — matrix operand order repaired and checked by reading, not compiled, not run on hardware
//
#include <hip/hip_runtime.h>

typedef __attribute__((ext_vector_type(16))) _Float16 v16h;
typedef __attribute__((ext_vector_type(8)))  float    v8f;

#define Bsz   512
#define Npts  1024
#define Ksel  8
#define Pper  2016
#define H1    128
#define H2    64
#define NCLS  40
#define NC3   48               // NCLS padded to 3 x 16
#define NTILES (Pper / 16)     // 126 row tiles of 16

// Position (in halfs) of element (row, k) inside one 512-half A-fragment chunk
// (this file's contiguous-K 16x32 f16 A layout, not the ISA table order):
//   lanes 0-15: M=row, K = {0..15} in halfs 0..15
//   lanes 16-31: M=row, K = {16..31} in halfs 0..15
__device__ __forceinline__ int a_pos(int row, int k) {
  int lane = (row & 15) | (k & 16);
  int slot = k & 15;
  return lane * 16 + slot;
}

__global__ __launch_bounds__(128)
void fused_sym_mlp(const float* __restrict__ x, const int* __restrict__ idx,
                   const int* __restrict__ perms,
                   const float* __restrict__ W1, const float* __restrict__ b1,
                   const float* __restrict__ W2, const float* __restrict__ b2,
                   const float* __restrict__ W3, const float* __restrict__ b3,
                   float* __restrict__ out)
{
  // Weights pre-swizzled into B-fragment layout (32x16 f16 B-matrix layout:
  // lane l holds column l&15, K = (l>=16?16:0)+h for half h -> contiguous K stripe).
  __shared__ __align__(32) _Float16 sW1[8 * 512];        // 8 ntiles,   K 0..31 (24 + pad)
  __shared__ __align__(32) _Float16 sW2[16 * 512];       // 4 ntiles x 4 kchunks
  __shared__ __align__(32) _Float16 sW3[6 * 512];        // 3 ntiles x 2 kchunks
  __shared__ __align__(32) _Float16 sA [4][512];         // per-wave input tile (A layout)
  __shared__ __align__(32) _Float16 sH1[4][4 * 512];     // per-wave h1 bounce (A layout)
  __shared__ __align__(32) _Float16 sH2[4][2 * 512];     // per-wave h2 bounce (A layout)
  __shared__ float sb1[H1];
  __shared__ float sb2[H2];
  __shared__ float spts[Ksel][4];                        // 8 gathered points (xyz + pad)
  __shared__ float yacc[NC3];

  const int b    = blockIdx.x;
  const int tid  = threadIdx.x;
  const int wave = tid >> 5;
  const int lane = tid & 31;

  // ---- one-time staging: f32 -> f16 weight fragments in LDS ----
  for (int i = tid; i < 8 * 512; i += 128) {             // W1: (24->32) x 128
    int t = i >> 9, l = (i >> 4) & 31, h = i & 15;
    int k = ((l & 16) ? 16 : 0) + h, c = t * 16 + (l & 15);
    sW1[i] = (_Float16)((k < 24) ? W1[k * H1 + c] : 0.f);
  }
  for (int i = tid; i < 16 * 512; i += 128) {            // W2: 128 x 64
    int t = i >> 11, kc = (i >> 9) & 3, l = (i >> 4) & 31, h = i & 15;
    int k = kc * 32 + ((l & 16) ? 16 : 0) + h, c = t * 16 + (l & 15);
    sW2[i] = (_Float16)W2[k * H2 + c];
  }
  for (int i = tid; i < 6 * 512; i += 128) {             // W3: 64 x (40->48)
    int t = i >> 10, kc = (i >> 9) & 1, l = (i >> 4) & 31, h = i & 15;
    int k = kc * 32 + ((l & 16) ? 16 : 0) + h, c = t * 16 + (l & 15);
    sW3[i] = (_Float16)((c < NCLS) ? W3[k * NCLS + c] : 0.f);
  }
  if (tid < H1)  sb1[tid]  = b1[tid];
  if (tid < H2)  sb2[tid]  = b2[tid];
  if (tid < NC3) yacc[tid] = 0.f;
  if (tid < Ksel * 3) {                                  // gather this batch's 8 points
    int j = tid / 3, c = tid % 3;
    spts[j][c] = x[(size_t)b * (Npts * 3) + (size_t)idx[j] * 3 + c];
  }
  __syncthreads();

  float accY0 = 0.f, accY1 = 0.f, accY2 = 0.f;

  for (int tile = wave; tile < NTILES; tile += 4) {
    const int p0 = tile * 16;

    // ---- build 16x32 f16 A tile from points + perms (K 24..31 zero pad) ----
    #pragma unroll
    for (int it = 0; it < 4; ++it) {
      int i = lane + it * 32;                            // 0..127 -> (row, slot)
      int row = i >> 3, slot = i & 7;
      int j = perms[((size_t)b * Pper + (p0 + row)) * Ksel + slot];
      #pragma unroll
      for (int c = 0; c < 3; ++c)
        sA[wave][a_pos(row, slot * 3 + c)] = (_Float16)spts[j][c];
      sA[wave][a_pos(row, 24 + slot)] = (_Float16)0.f;
    }
    __builtin_amdgcn_fence(__ATOMIC_ACQ_REL, "wavefront");

    v16h aIn = *(const v16h*)&sA[wave][lane * 16];

    // ---- layer 1: (16x32) @ (32x128), bias + ReLU, repack to A layout ----
    #pragma unroll
    for (int t = 0; t < 8; ++t) {
      v16h bf = *(const v16h*)&sW1[t * 512 + lane * 16];
      v8f c = {};
      c = __builtin_amdgcn_wmma_f32_16x16x32_f16(false, aIn, false, bf,
                                                 (short)0, c, false, false);
      float bias = sb1[t * 16 + (lane & 15)];
      #pragma unroll
      for (int r = 0; r < 8; ++r) {
        float v = c[r] + bias;
        v = v > 0.f ? v : 0.f;
        int row = r + ((lane & 16) ? 8 : 0);
        int k   = t * 16 + (lane & 15);
        sH1[wave][(k >> 5) * 512 + a_pos(row, k & 31)] = (_Float16)v;
      }
    }
    __builtin_amdgcn_fence(__ATOMIC_ACQ_REL, "wavefront");

    v16h h1f[4];
    #pragma unroll
    for (int kc = 0; kc < 4; ++kc)
      h1f[kc] = *(const v16h*)&sH1[wave][kc * 512 + lane * 16];

    // ---- layer 2: (16x128) @ (128x64), bias + ReLU, repack ----
    #pragma unroll
    for (int t = 0; t < 4; ++t) {
      v8f c = {};
      #pragma unroll
      for (int kc = 0; kc < 4; ++kc) {
        v16h bf = *(const v16h*)&sW2[(t * 4 + kc) * 512 + lane * 16];
        c = __builtin_amdgcn_wmma_f32_16x16x32_f16(false, h1f[kc], false, bf,
                                                   (short)0, c, false, false);
      }
      float bias = sb2[t * 16 + (lane & 15)];
      #pragma unroll
      for (int r = 0; r < 8; ++r) {
        float v = c[r] + bias;
        v = v > 0.f ? v : 0.f;
        int row = r + ((lane & 16) ? 8 : 0);
        int k   = t * 16 + (lane & 15);
        sH2[wave][(k >> 5) * 512 + a_pos(row, k & 31)] = (_Float16)v;
      }
    }
    __builtin_amdgcn_fence(__ATOMIC_ACQ_REL, "wavefront");

    v16h h2f[2];
    #pragma unroll
    for (int kc = 0; kc < 2; ++kc)
      h2f[kc] = *(const v16h*)&sH2[wave][kc * 512 + lane * 16];

    // ---- layer 3: (16x64) @ (64x48); reduce rows into per-lane accumulators ----
    #pragma unroll
    for (int t = 0; t < 3; ++t) {
      v8f c = {};
      #pragma unroll
      for (int kc = 0; kc < 2; ++kc) {
        v16h bf = *(const v16h*)&sW3[(t * 2 + kc) * 512 + lane * 16];
        c = __builtin_amdgcn_wmma_f32_16x16x32_f16(false, h2f[kc], false, bf,
                                                   (short)0, c, false, false);
      }
      float s = 0.f;
      #pragma unroll
      for (int r = 0; r < 8; ++r) s += c[r];             // sum over 8 rows this lane holds
      if      (t == 0) accY0 += s;
      else if (t == 1) accY1 += s;
      else             accY2 += s;
    }
    __builtin_amdgcn_fence(__ATOMIC_ACQ_REL, "wavefront"); // WAR vs next tile's sA/sH writes
  }

  // ---- reduce over waves/lane-halves; mean over P; add b3 ----
  atomicAdd(&yacc[ 0 + (lane & 15)], accY0);
  atomicAdd(&yacc[16 + (lane & 15)], accY1);
  atomicAdd(&yacc[32 + (lane & 15)], accY2);
  __syncthreads();
  if (tid < NCLS)
    out[(size_t)b * NCLS + tid] = yacc[tid] * (1.f / (float)Pper) + b3[tid];
}

extern "C" void kernel_launch(void* const* d_in, const int* in_sizes, int n_in,
                              void* d_out, int out_size, void* d_ws, size_t ws_size,
                              hipStream_t stream) {
  const float* x    = (const float*)d_in[0];
  const int*   idx  = (const int*)  d_in[1];
  const int*   prm  = (const int*)  d_in[2];
  const float* W1   = (const float*)d_in[3];
  const float* b1   = (const float*)d_in[4];
  const float* W2   = (const float*)d_in[5];
  const float* b2   = (const float*)d_in[6];
  const float* W3   = (const float*)d_in[7];
  const float* b3   = (const float*)d_in[8];
  fused_sym_mlp<<<dim3(Bsz), dim3(128), 0, stream>>>(x, idx, prm, W1, b1, W2, b2, W3, b3,
                                                     (float*)d_out);
}
